// SelfAttention_32203664785949
// MI455X (gfx1250) — compile-verified, matrix operand order checked by reading, not run on hardware
//
#include <hip/hip_runtime.h>
#include <hip/hip_bf16.h>

// ---------------------------------------------------------------------------
// Self-attention forward on gfx1250 (MI455X), bf16 WMMA with f32 accumulation.
//   x:[4,2048,1024] f32, w_qkv:[1024,3072] f32, b_qkv:[3072] f32,
//   w_out:[1024,1024] f32, b_out:[1024] f32  ->  out:[4,2048,1024] f32
// Data movement: TDM tensor_load_to_lds (K tiles), global_load_async_to_lds
// (out-proj A tiles), global_prefetch (next GEMM tiles).
// ---------------------------------------------------------------------------

#define DIM    1024
#define NH     16
#define HD     64
#define SEQ    2048
#define BATCH  4
#define M_TOT  (BATCH * SEQ)   // 8192
#define N_QKV  (3 * DIM)       // 3072

typedef __attribute__((ext_vector_type(8)))  unsigned short v8u;
typedef __attribute__((ext_vector_type(16))) unsigned short v16u;
typedef __attribute__((ext_vector_type(2)))  __bf16        v2bf;
typedef __attribute__((ext_vector_type(16))) __bf16        v16bf;
typedef __attribute__((ext_vector_type(8)))  float         v8f;
typedef __attribute__((ext_vector_type(4)))  unsigned int  v4ui;
typedef __attribute__((ext_vector_type(8)))  int           v8i;
typedef __attribute__((ext_vector_type(4)))  int           v4i;

union FragU {
    struct { v8u lo; v8u hi; } p;
    v16u u;
    v16bf b;
};

__device__ __forceinline__ unsigned short f2bf(float f) {
    unsigned int u = __float_as_uint(f);
    u += 0x7FFFu + ((u >> 16) & 1u);     // round-to-nearest-even
    return (unsigned short)(u >> 16);
}

// packed f32x2 -> bf16x2 (v_cvt_pk_bf16_f32 when available)
__device__ __forceinline__ unsigned int f2bf2(float lo, float hi) {
#if __has_builtin(__builtin_amdgcn_cvt_pk_bf16_f32)
    union { v2bf v; unsigned int u; } c;
    c.v = __builtin_amdgcn_cvt_pk_bf16_f32(lo, hi);
    return c.u;
#else
    return (unsigned int)f2bf(lo) | ((unsigned int)f2bf(hi) << 16);
#endif
}

// Low 32 bits of a generic pointer to __shared__ = LDS byte address.
__device__ __forceinline__ unsigned lds_addr32(const void* p) {
    return (unsigned)(unsigned long long)p;
}

// Async 16B copy global -> LDS (tracked with ASYNCcnt). saddr form:
//   mem = SGPR_base + voffset ; LDS[vdst] = mem[0..15]
__device__ __forceinline__ void async_copy_b128(unsigned lds_byte_addr,
                                                const void* sbase,
                                                unsigned voff_bytes) {
    asm volatile("global_load_async_to_lds_b128 %0, %1, %2"
                 :: "v"(lds_byte_addr), "v"(voff_bytes),
                    "s"((unsigned long long)sbase)
                 : "memory");
}
__device__ __forceinline__ void wait_async() {
    asm volatile("s_wait_asynccnt 0" ::: "memory");
}

__device__ __forceinline__ void wait_tensor() {
#if __has_builtin(__builtin_amdgcn_s_wait_tensorcnt)
    __builtin_amdgcn_s_wait_tensorcnt(0);
#else
    asm volatile("s_wait_tensorcnt 0" ::: "memory");
#endif
}

#if __has_builtin(__builtin_amdgcn_tensor_load_to_lds)
#define HAVE_TDM 1
// TDM: DMA a 2D tile of bf16 (tileW elems x tileH rows, global row stride
// strideElems) into LDS at ldsAddr. LDS rows are padded by 4 DWORDs (16B)
// after every 32 DWORDs (128B row) -> pad_interval=4, pad_amount=3 per D#.
// Whole tile moved by ONE instruction from one wave; tracked by TENSORcnt.
__device__ __forceinline__ void tdm_load_2d(unsigned ldsAddr, const void* gptr,
                                            unsigned tileW, unsigned tileH,
                                            unsigned strideElems) {
    unsigned long long ga = (unsigned long long)gptr;
    v4ui g0;
    g0.x = 1u;                                            // count=1 (valid D#)
    g0.y = ldsAddr;                                       // lds_addr [63:32]
    g0.z = (unsigned)ga;                                  // global_addr lo
    g0.w = (unsigned)((ga >> 32) & 0x1FFFFFFu) | (2u << 30); // addr hi + type=2
    v8i g1;
    g1[0] = (int)((1u << 16) |                            // data_size = 2B
                  (1u << 20) |                            // pad_enable
                  (4u << 22) |                            // pad_interval: 32 dw
                  (3u << 25));                            // pad_amount: 4 dw
    g1[1] = (int)(tileW << 16);                           // tensor_dim0 lo16
    g1[2] = (int)(tileH << 16);                           // dim0 hi=0 | dim1 lo16
    g1[3] = (int)(tileW << 16);                           // dim1 hi=0 | tile_dim0
    g1[4] = (int)tileH;                                   // tile_dim1 | tile_dim2=0
    g1[5] = (int)strideElems;                             // tensor_dim0_stride lo32
    g1[6] = 0;                                            // stride hi | dim1_stride lo
    g1[7] = 0;
    v4i gz = {0, 0, 0, 0};
#if defined(__clang_major__) && __clang_major__ >= 23
    v8i gz8 = {0, 0, 0, 0, 0, 0, 0, 0};
    __builtin_amdgcn_tensor_load_to_lds(g0, g1, gz, gz, gz8, 0);
#else
    __builtin_amdgcn_tensor_load_to_lds(g0, g1, gz, gz, 0);
#endif
}
#else
#define HAVE_TDM 0
#endif

// Load a 16x32 bf16 A/B fragment per the CDNA5 16-bit layout:
// lane holds row (A: M, B with col-major storage: N) = `row`,
// K elements [kbase + lh*8 .. +7] and [kbase + 16 + lh*8 .. +7].
__device__ __forceinline__ v16bf ld_frag(const unsigned short* base, int row,
                                         int stride, int kbase, int lh) {
    FragU f;
    const unsigned short* p = base + (row * stride + kbase + lh * 8);
    f.p.lo = *(const v8u*)(p);
    f.p.hi = *(const v8u*)(p + 16);
    return f.b;
}

__device__ __forceinline__ v8f wmma_bf16(v16bf a, v16bf b, v8f c) {
    return __builtin_amdgcn_wmma_f32_16x16x32_bf16(
        /*neg_a=*/false, a, /*neg_b=*/false, b,
        /*c_mod=*/(short)0, c, /*reuse_a=*/false, /*reuse_b=*/false);
}

// ---------------------------------------------------------------------------
// Kernel 1: QKV = x @ w_qkv + b_qkv   (f32 in, bf16 out, head-split layout)
// grid (24, 64), 256 threads. Block tile 128x128, BK=64.
// ---------------------------------------------------------------------------
__global__ __launch_bounds__(256) void qkv_gemm(
    const float* __restrict__ x, const float* __restrict__ wqkv,
    const float* __restrict__ bqkv,
    unsigned short* __restrict__ qb, unsigned short* __restrict__ kb,
    unsigned short* __restrict__ vb)
{
    __shared__ unsigned short ldsA[128 * 72];   // [m][k], padded
    __shared__ unsigned short ldsB[128 * 72];   // [n][k] (B stored col-major)

    const int tid = threadIdx.x;
    const int wid = tid >> 5, lane = tid & 31;
    const int ln = lane & 15, lh = lane >> 4;
    const int m0 = blockIdx.y * 128;
    const int n0 = blockIdx.x * 128;
    const int wm = (wid >> 2) * 64;             // 0 / 64
    const int wn = (wid & 3) * 32;              // 0..96

    v8f acc[4][2];
    const v8f vz = {};
#pragma unroll
    for (int i = 0; i < 4; ++i)
#pragma unroll
        for (int j = 0; j < 2; ++j) acc[i][j] = vz;

    for (int kk = 0; kk < DIM; kk += 64) {
        // uniform bases (SGPR) + 32-bit per-lane offsets -> saddr addressing
        const float* xb = x + (size_t)m0 * DIM + kk;
        const float* wb = wqkv + (size_t)kk * N_QKV + n0;

        __syncthreads();
        // stage A (f32 -> bf16), 128x64
#pragma unroll
        for (int i = 0; i < 8; ++i) {
            unsigned e = (unsigned)(tid + i * 256) * 4;
            unsigned m = e >> 6, k = e & 63;
            const float4 f = *(const float4*)(xb + (m * DIM + k));
            unsigned long long pk =
                (unsigned long long)f2bf2(f.x, f.y) |
                ((unsigned long long)f2bf2(f.z, f.w) << 32);
            *(unsigned long long*)(ldsA + (m * 72 + k)) = pk;
        }
        // stage B transposed (f32 -> bf16), [n][k]
#pragma unroll
        for (int i = 0; i < 8; ++i) {
            unsigned e = (unsigned)(tid + i * 256) * 4;
            unsigned k = e >> 7, n = e & 127;
            const float4 f = *(const float4*)(wb + (k * N_QKV + n));
            ldsB[(n + 0) * 72 + k] = f2bf(f.x);
            ldsB[(n + 1) * 72 + k] = f2bf(f.y);
            ldsB[(n + 2) * 72 + k] = f2bf(f.z);
            ldsB[(n + 3) * 72 + k] = f2bf(f.w);
        }
        __syncthreads();

        if (kk + 64 < DIM) {   // prefetch next tiles -> global_prefetch_b8
            __builtin_prefetch(xb + 64 + (tid >> 1) * DIM + (tid & 1) * 32, 0, 1);
            __builtin_prefetch(wb + 64 * N_QKV + (tid >> 2) * N_QKV + (tid & 3) * 32, 0, 1);
        }

#pragma unroll
        for (int ks = 0; ks < 64; ks += 32) {
            v16bf bfr[2];
#pragma unroll
            for (int fn = 0; fn < 2; ++fn)
                bfr[fn] = ld_frag(ldsB, wn + fn * 16 + ln, 72, ks, lh);
#pragma unroll
            for (int fm = 0; fm < 4; ++fm) {
                v16bf af = ld_frag(ldsA, wm + fm * 16 + ln, 72, ks, lh);
#pragma unroll
                for (int fn = 0; fn < 2; ++fn)
                    acc[fm][fn] = wmma_bf16(af, bfr[fn], acc[fm][fn]);
            }
        }
    }

    // epilogue: bias + scatter bf16 into [b,h,n,hd] Q/K/V buffers
#pragma unroll
    for (int fm = 0; fm < 4; ++fm)
#pragma unroll
        for (int fn = 0; fn < 2; ++fn) {
            int c = n0 + wn + fn * 16 + ln;
            int which = c >> 10;               // 0=q 1=k 2=v
            int dcol = c & (DIM - 1);
            int h = dcol >> 6, d = dcol & 63;
            unsigned short* dst = (which == 0) ? qb : (which == 1) ? kb : vb;
            float bias = bqkv[c];
#pragma unroll
            for (int r = 0; r < 8; ++r) {
                int m = m0 + wm + fm * 16 + r + lh * 8;
                int bi = m >> 11, n = m & (SEQ - 1);
                unsigned off = ((((unsigned)bi * NH + h) * SEQ + n) << 6) + d;
                dst[off] = f2bf(acc[fm][fn][r] + bias);
            }
        }
}

// ---------------------------------------------------------------------------
// Kernel 2: flash attention per (b*h). grid (16, 64), 256 threads.
// Block: 128 query rows (16 per wave); K/V tiles of 128 keys in LDS.
// K tile staged by ONE tensor_load_to_lds DMA (TDM, TENSORcnt path).
// ---------------------------------------------------------------------------
__global__ __launch_bounds__(256) void attn_kernel(
    const unsigned short* __restrict__ qb, const unsigned short* __restrict__ kb,
    const unsigned short* __restrict__ vb, unsigned short* __restrict__ aw)
{
    __shared__ unsigned short ldsK[128 * 72];        // [key][d]
    __shared__ unsigned short ldsV[64 * 136];        // [d][key] (transposed)
    __shared__ unsigned short ldsP[8 * 16 * 136];    // per-wave P patch

    const int tid = threadIdx.x;
    const int wid = tid >> 5, lane = tid & 31;
    const int ln = lane & 15, lh = lane >> 4;
    const int bh = blockIdx.y;                       // b*16 + h
    const int qtile = blockIdx.x * 128;
    const size_t base = (size_t)bh * SEQ * HD;
    const unsigned short* Q = qb + base;
    const unsigned short* K = kb + base;
    const unsigned short* V = vb + base;
    const int qrow = qtile + wid * 16;
    const unsigned kLds = lds_addr32(ldsK);

    // Q 16x64 tile as two A fragments, kept in registers.
    v16bf qa[2];
#pragma unroll
    for (int f = 0; f < 2; ++f)
        qa[f] = ld_frag(Q, qrow + ln, HD, f * 32, lh);

    const v8f vz = {};
    v8f o[4];
#pragma unroll
    for (int i = 0; i < 4; ++i) o[i] = vz;
    float mrow[8], lsum[8];
#pragma unroll
    for (int r = 0; r < 8; ++r) { mrow[r] = -1e30f; lsum[r] = 0.f; }
    const float scale = 0.125f;                      // hd^-0.5 = 1/8

    for (int kt = 0; kt < SEQ; kt += 128) {
        const unsigned short* Kt = K + kt * HD;
        const unsigned short* Vt = V + kt * HD;

        __syncthreads();
#if HAVE_TDM
        // one DMA moves the whole 128x64 bf16 K tile, LDS-padded to stride 72
        if (wid == 0)
            tdm_load_2d(kLds, Kt, HD, 128, HD);
#else
#pragma unroll
        for (int i = 0; i < 4; ++i) {
            int c = tid + i * 256;
            int key = c >> 3, part = (c & 7) * 8;
            async_copy_b128(kLds + (unsigned)((key * 72 + part) * 2),
                            Kt, (unsigned)((key * HD + part) * 2));
        }
#endif
        // stage V tile transposed -> [d][key]
#pragma unroll
        for (int i = 0; i < 16; ++i) {
            int u = tid + i * 256;
            int key = u >> 5, d2 = (u & 31) * 2;
            unsigned int w = *(const unsigned int*)(Vt + (key * HD + d2));
            ldsV[(d2 + 0) * 136 + key] = (unsigned short)(w & 0xFFFFu);
            ldsV[(d2 + 1) * 136 + key] = (unsigned short)(w >> 16);
        }
#if HAVE_TDM
        if (wid == 0) wait_tensor();
#else
        wait_async();
#endif
        __syncthreads();

        // S = Q @ K^T  (16 x 128), 8 frags x 2 K-steps
        v8f s[8];
#pragma unroll
        for (int i = 0; i < 8; ++i) s[i] = vz;
#pragma unroll
        for (int ks = 0; ks < 64; ks += 32) {
            v16bf af = qa[ks >> 5];
#pragma unroll
            for (int fn = 0; fn < 8; ++fn) {
                v16bf bf = ld_frag(ldsK, fn * 16 + ln, 72, ks, lh);
                s[fn] = wmma_bf16(af, bf, s[fn]);
            }
        }

        // scale + online softmax
#pragma unroll
        for (int fn = 0; fn < 8; ++fn)
#pragma unroll
            for (int r = 0; r < 8; ++r) s[fn][r] *= scale;

        float mnew[8], alpha[8];
#pragma unroll
        for (int r = 0; r < 8; ++r) {
            float v = mrow[r];
#pragma unroll
            for (int fn = 0; fn < 8; ++fn) v = fmaxf(v, s[fn][r]);
#pragma unroll
            for (int msk = 1; msk < 16; msk <<= 1)
                v = fmaxf(v, __shfl_xor(v, msk, 32));
            mnew[r] = v;
            alpha[r] = __expf(mrow[r] - v);
        }
#pragma unroll
        for (int fo = 0; fo < 4; ++fo)
#pragma unroll
            for (int r = 0; r < 8; ++r) o[fo][r] *= alpha[r];

        unsigned short* P = ldsP + wid * (16 * 136);
        float rsum[8];
#pragma unroll
        for (int r = 0; r < 8; ++r) rsum[r] = 0.f;
#pragma unroll
        for (int fn = 0; fn < 8; ++fn)
#pragma unroll
            for (int r = 0; r < 8; ++r) {
                float p = __expf(s[fn][r] - mnew[r]);   // v_exp_f32 co-exec
                rsum[r] += p;
                P[(r + lh * 8) * 136 + fn * 16 + ln] = f2bf(p);
            }
#pragma unroll
        for (int r = 0; r < 8; ++r) {
            float v = rsum[r];
#pragma unroll
            for (int msk = 1; msk < 16; msk <<= 1) v += __shfl_xor(v, msk, 32);
            lsum[r] = lsum[r] * alpha[r] + v;
            mrow[r] = mnew[r];
        }

        // make per-wave P writes visible to all lanes of this wave
        __builtin_amdgcn_wave_barrier();
        asm volatile("s_wait_dscnt 0" ::: "memory");
        __builtin_amdgcn_wave_barrier();

        // O += P @ V   (16 x 64), K = 128 -> 4 steps
#pragma unroll
        for (int ks = 0; ks < 128; ks += 32) {
            v16bf pa = ld_frag(P, ln, 136, ks, lh);
#pragma unroll
            for (int fo = 0; fo < 4; ++fo) {
                v16bf bf = ld_frag(ldsV, fo * 16 + ln, 136, ks, lh);
                o[fo] = wmma_bf16(pa, bf, o[fo]);
            }
        }
    }

    // finalize and write bf16 attention output as [b*n, h*64+d]
    const int b = bh >> 4, h = bh & 15;
#pragma unroll
    for (int r = 0; r < 8; ++r) {
        float inv = 1.0f / lsum[r];
        int row = b * SEQ + qrow + r + lh * 8;
#pragma unroll
        for (int fo = 0; fo < 4; ++fo)
            aw[row * DIM + h * 64 + fo * 16 + ln] = f2bf(o[fo][r] * inv);
    }
}

// ---------------------------------------------------------------------------
// Kernel 3: out = attn(bf16) @ w_out + b_out   (f32 out)
// grid (8, 64), 256 threads, tile 128x128, BK=64. A staged via async-LDS.
// ---------------------------------------------------------------------------
__global__ __launch_bounds__(256) void outproj_gemm(
    const unsigned short* __restrict__ aw, const float* __restrict__ wout,
    const float* __restrict__ bout, float* __restrict__ out)
{
    __shared__ unsigned short ldsA[128 * 72];
    __shared__ unsigned short ldsB[128 * 72];

    const int tid = threadIdx.x;
    const int wid = tid >> 5, lane = tid & 31;
    const int ln = lane & 15, lh = lane >> 4;
    const int m0 = blockIdx.y * 128;
    const int n0 = blockIdx.x * 128;
    const int wm = (wid >> 2) * 64;
    const int wn = (wid & 3) * 32;
    const unsigned aLds = lds_addr32(ldsA);

    v8f acc[4][2];
    const v8f vz = {};
#pragma unroll
    for (int i = 0; i < 4; ++i)
#pragma unroll
        for (int j = 0; j < 2; ++j) acc[i][j] = vz;

    for (int kk = 0; kk < DIM; kk += 64) {
        const unsigned short* awb = aw + (size_t)m0 * DIM + kk;
        const float* wb = wout + (size_t)kk * DIM + n0;

        __syncthreads();
        // stage A (already bf16): async global->LDS 16B copies (ASYNCcnt path)
#pragma unroll
        for (int i = 0; i < 4; ++i) {
            int c = tid + i * 256;
            int m = c >> 3, part = (c & 7) * 8;
            async_copy_b128(aLds + (unsigned)((m * 72 + part) * 2),
                            awb, (unsigned)((m * DIM + part) * 2));
        }
        // stage B (f32 -> bf16, transposed [n][k])
#pragma unroll
        for (int i = 0; i < 8; ++i) {
            unsigned e = (unsigned)(tid + i * 256) * 4;
            unsigned k = e >> 7, n = e & 127;
            const float4 f = *(const float4*)(wb + (k * DIM + n));
            ldsB[(n + 0) * 72 + k] = f2bf(f.x);
            ldsB[(n + 1) * 72 + k] = f2bf(f.y);
            ldsB[(n + 2) * 72 + k] = f2bf(f.z);
            ldsB[(n + 3) * 72 + k] = f2bf(f.w);
        }
        wait_async();
        __syncthreads();

        if (kk + 64 < DIM) {
            __builtin_prefetch(awb + 64 + (tid >> 2) * DIM + (tid & 3) * 16, 0, 1);
            __builtin_prefetch(wb + 64 * DIM + (tid >> 2) * DIM + (tid & 3) * 32, 0, 1);
        }

#pragma unroll
        for (int ks = 0; ks < 64; ks += 32) {
            v16bf bfr[2];
#pragma unroll
            for (int fn = 0; fn < 2; ++fn)
                bfr[fn] = ld_frag(ldsB, wn + fn * 16 + ln, 72, ks, lh);
#pragma unroll
            for (int fm = 0; fm < 4; ++fm) {
                v16bf af = ld_frag(ldsA, wm + fm * 16 + ln, 72, ks, lh);
#pragma unroll
                for (int fn = 0; fn < 2; ++fn)
                    acc[fm][fn] = wmma_bf16(af, bfr[fn], acc[fm][fn]);
            }
        }
    }

#pragma unroll
    for (int fm = 0; fm < 4; ++fm)
#pragma unroll
        for (int fn = 0; fn < 2; ++fn) {
            int c = n0 + wn + fn * 16 + ln;
            float bias = bout[c];
#pragma unroll
            for (int r = 0; r < 8; ++r) {
                int m = m0 + wm + fm * 16 + r + lh * 8;
                out[m * DIM + c] = acc[fm][fn][r] + bias;
            }
        }
}

// ---------------------------------------------------------------------------
extern "C" void kernel_launch(void* const* d_in, const int* in_sizes, int n_in,
                              void* d_out, int out_size, void* d_ws, size_t ws_size,
                              hipStream_t stream) {
    (void)in_sizes; (void)n_in; (void)out_size; (void)ws_size;
    const float* x    = (const float*)d_in[0];
    const float* wqkv = (const float*)d_in[1];
    const float* bqkv = (const float*)d_in[2];
    const float* wout = (const float*)d_in[3];
    const float* bout = (const float*)d_in[4];
    float* out = (float*)d_out;

    const size_t plane = (size_t)M_TOT * DIM;        // 8M bf16 elements
    unsigned short* qb = (unsigned short*)d_ws;
    unsigned short* kb = qb + plane;
    unsigned short* vb = kb + plane;
    unsigned short* aw = vb + plane;                 // total 64 MB of ws

    qkv_gemm    <<<dim3(N_QKV / 128, M_TOT / 128), 256, 0, stream>>>(x, wqkv, bqkv, qb, kb, vb);
    attn_kernel <<<dim3(SEQ / 128, BATCH * NH),    256, 0, stream>>>(qb, kb, vb, aw);
    outproj_gemm<<<dim3(DIM / 128, M_TOT / 128),   256, 0, stream>>>(aw, wout, bout, out);
}
